// RFMNet11_75917841924418
// MI455X (gfx1250) — matrix operand order repaired and checked by reading, not compiled, not run on hardware
//
#include <hip/hip_runtime.h>

// ---------------------------------------------------------------------------
// RFMNet / DiffusionNet-style forward for MI455X (gfx1250, wave32, WMMA).
// GEMMs: v_wmma_f32_16x16x32_bf16. Tiles are DMA'd into LDS by the Tensor
// Data Mover (tensor_load_to_lds, TENSORcnt), double-buffered; fragments are
// contiguous ds_load_b128 reads per ISA 7.12.2 (B operands stored col-major).
// ---------------------------------------------------------------------------

typedef __attribute__((ext_vector_type(16))) __bf16 v16bf;
typedef __attribute__((ext_vector_type(8)))  float  v8f;
typedef __attribute__((ext_vector_type(4)))  unsigned int u32x4;
typedef __attribute__((ext_vector_type(8)))  int i32x8;
typedef __attribute__((ext_vector_type(4)))  int i32x4;
typedef __attribute__((address_space(3))) unsigned char lds_u8_t;

#define BB    2
#define VV    4096
#define KK    128
#define CIN_  16
#define COUT_ 16
#define CW_   128
#define NB_   4

__device__ __forceinline__ unsigned short f2bf(float f) {
  unsigned u = __float_as_uint(f);
  u += 0x7fffu + ((u >> 16) & 1u);          // round-to-nearest-even
  return (unsigned short)(u >> 16);
}

__device__ __forceinline__ unsigned lds_addr(void* p) {
  // generic -> LDS addrspace(3) -> raw 32-bit LDS byte offset
  return (unsigned)(unsigned long long)(lds_u8_t*)p;
}

// TDM: DMA a 2D tile (tileRows x tileDw dwords, row stride strideDw dwords)
// from global memory into LDS. D# per CDNA5 ISA sec 8.3/8.4:
//   group0: count=1 | lds_addr[63:32] | global_addr[120:64] | type=2
//   group1: data_size=4B, tensor dims huge (no OOB clipping), tile_dim0/1,
//           tensor_dim0_stride; groups 2/3 zero (<=2D tensor).
// This toolchain's builtin is the 6-arg form:
//   (u32x4 g0, i32x8 g1, i32x4 g2, i32x4 g3, i32x8 g4, i32 cpol)
__device__ __forceinline__ void tdm_load_2d(unsigned ldsByte, const void* g,
                                            unsigned tileDw, unsigned tileRows,
                                            unsigned strideDw) {
  unsigned long long ga = (unsigned long long)g;
  u32x4 g0;
  g0.x = 1u;                                   // count=1 (valid user descriptor)
  g0.y = ldsByte;                              // lds_addr
  g0.z = (unsigned)ga;                         // global_addr[31:0]
  g0.w = (unsigned)(ga >> 32) | (2u << 30);    // global_addr[56:32] | type=2
  const unsigned td0 = 0x7fffffffu, td1 = 0x7fffffffu;
  i32x8 g1;
  g1[0] = (int)(2u << 16);                               // wg_mask=0, data_size=4B
  g1[1] = (int)((td0 & 0xffffu) << 16);                  // tensor_dim0 lo16
  g1[2] = (int)((td0 >> 16) | ((td1 & 0xffffu) << 16));  // dim0 hi | dim1 lo
  g1[3] = (int)((td1 >> 16) | (tileDw << 16));           // dim1 hi | tile_dim0
  g1[4] = (int)(tileRows & 0xffffu);                     // tile_dim1 | tile_dim2=0
  g1[5] = (int)strideDw;                                 // tensor_dim0_stride[31:0]
  g1[6] = 0;                                             // stride hi | dim1_stride lo
  g1[7] = 0;
  i32x4 z4 = {0, 0, 0, 0};
  i32x8 z8 = {0, 0, 0, 0, 0, 0, 0, 0};
  __builtin_amdgcn_tensor_load_to_lds(g0, g1, z4, z4, z8, 0);
}

union FragU { uint4 q[2]; v16bf v; };

// ---------------------------------------------------------------------------
// C(M x 128) = A(M x K, row-major, lda) * Bt(128 x K col-major view, ldbT)
// K must be a multiple of 64. Block tile 128x128, K-step 32, 8 waves,
// each wave = 2x4 WMMA 16x16 tiles. Tiles staged by TDM, double-buffered.
// ---------------------------------------------------------------------------
template <bool HAS_C, bool HAS_CB, bool HAS_CBT, bool HAS_BIAS, bool HAS_RESID, bool RELU>
__global__ __launch_bounds__(256)
void gemm_bf16(const unsigned short* __restrict__ A, long long sAb, int lda,
               const unsigned short* __restrict__ Bt, long long sBb, int ldbT,
               float* __restrict__ C, long long sCb, int ldc,
               unsigned short* __restrict__ Cb, long long sCbb, int ldcb,
               unsigned short* __restrict__ CbT, long long sTb, int ldT,
               const float* __restrict__ bias,
               const float* __restrict__ resid, long long sRb, int ldr,
               int M, int K)
{
  __shared__ __align__(16) unsigned short lA[2][128 * 32];
  __shared__ __align__(16) unsigned short lB[2][128 * 32];

  const int bz = blockIdx.z;
  const unsigned short* Ap = A  + (long long)bz * sAb + (long long)blockIdx.x * 128 * lda;
  const unsigned short* Bp = Bt + (long long)bz * sBb;

  const int tid  = threadIdx.x;
  const int wave = tid >> 5, lane = tid & 31;
  const int hl   = lane >> 4, m16 = lane & 15;
  const int wm   = wave & 3;   // 0..3 : 32-row region
  const int wn   = wave >> 2;  // 0..1 : 64-col region

  unsigned aB[2], bB[2];
  #pragma unroll
  for (int bf = 0; bf < 2; bf++) {
    aB[bf] = lds_addr(&lA[bf][0]);
    bB[bf] = lds_addr(&lB[bf][0]);
  }

  // One TDM descriptor per tile: 128 rows x 16 dwords, row stride ld/2 dwords.
  auto issue = [&](int bf, int k0) {
    if (tid < 32) {
      tdm_load_2d(aB[bf], Ap + k0, 16u, 128u, (unsigned)(lda  >> 1));
      tdm_load_2d(bB[bf], Bp + k0, 16u, 128u, (unsigned)(ldbT >> 1));
    }
  };

  v8f acc[2][4];
  #pragma unroll
  for (int i = 0; i < 2; i++)
    #pragma unroll
    for (int j = 0; j < 4; j++) {
      v8f z = {0.f, 0.f, 0.f, 0.f, 0.f, 0.f, 0.f, 0.f};
      acc[i][j] = z;
    }

  auto compute = [&](int cur) {
    // A fragment: two contiguous 8-elem K runs (hl*8 and 16+hl*8).
    v16bf af[2];
    #pragma unroll
    for (int tm = 0; tm < 2; tm++) {
      const unsigned short* base = &lA[cur][(wm * 32 + tm * 16 + m16) * 32 + hl * 8];
      FragU fa;
      fa.q[0] = *(const uint4*)(base);
      fa.q[1] = *(const uint4*)(base + 16);
      af[tm] = fa.v;
    }
    // B fragment: two contiguous 8-elem K runs (hl*8 and 16+hl*8), col-major tile.
    v16bf bfr[4];
    #pragma unroll
    for (int tn = 0; tn < 4; tn++) {
      const unsigned short* base = &lB[cur][(wn * 64 + tn * 16 + m16) * 32 + hl * 8];
      FragU fb;
      fb.q[0] = *(const uint4*)(base);
      fb.q[1] = *(const uint4*)(base + 16);
      bfr[tn] = fb.v;
    }
    #pragma unroll
    for (int tm = 0; tm < 2; tm++)
      #pragma unroll
      for (int tn = 0; tn < 4; tn++)
        acc[tm][tn] = __builtin_amdgcn_wmma_f32_16x16x32_bf16(
            false, af[tm], false, bfr[tn], (short)0, acc[tm][tn], false, false);
  };

  // Two-stage software pipeline; buffer index is a literal in each stage.
  issue(0, 0);
  for (int k0 = 0; k0 < K; k0 += 64) {
    // buffer 0 holds chunk k0; chunk k0+32 always exists (K % 64 == 0)
    issue(1, k0 + 32);
    if (tid < 32) __builtin_amdgcn_s_wait_tensorcnt(2);
    __syncthreads();
    compute(0);
    __syncthreads();

    // buffer 1 holds chunk k0+32
    const bool more = (k0 + 64) < K;
    if (more) { issue(0, k0 + 64); if (tid < 32) __builtin_amdgcn_s_wait_tensorcnt(2); }
    else      { if (tid < 32) __builtin_amdgcn_s_wait_tensorcnt(0); }
    __syncthreads();
    compute(1);
    __syncthreads();
  }

  // Epilogue: D layout — VGPR r: M = r + 8*half, N = lane%16.
  #pragma unroll
  for (int tm = 0; tm < 2; tm++) {
    #pragma unroll
    for (int tn = 0; tn < 4; tn++) {
      #pragma unroll
      for (int r = 0; r < 8; r++) {
        int gm = blockIdx.x * 128 + wm * 32 + tm * 16 + r + hl * 8;
        int gn = wn * 64 + tn * 16 + m16;
        float v = acc[tm][tn][r];
        if constexpr (HAS_BIAS)  v += bias[gn];
        if constexpr (HAS_RESID) v += resid[(long long)bz * sRb + (long long)gm * ldr + gn];
        if constexpr (RELU)      v = fmaxf(v, 0.f);
        if constexpr (HAS_C)   C  [(long long)bz * sCb  + (long long)gm * ldc  + gn] = v;
        if constexpr (HAS_CB)  Cb [(long long)bz * sCbb + (long long)gm * ldcb + gn] = f2bf(v);
        if constexpr (HAS_CBT) CbT[(long long)bz * sTb  + (long long)gn * ldT  + gm] = f2bf(v);
      }
    }
  }
}

// ------------------------- elementwise / setup kernels ----------------------

__global__ void k_cvt(const float* __restrict__ in, unsigned short* __restrict__ out, long long n) {
  long long i = (long long)blockIdx.x * 256 + threadIdx.x;
  if (i < n) out[i] = f2bf(in[i]);
}

// evecs (B,V,K) -> evecsT bf16 (B,K,V)
__global__ void k_transpose_cvt(const float* __restrict__ evecs, unsigned short* __restrict__ eT, int n) {
  int i = blockIdx.x * 256 + threadIdx.x;  // over B*V*K
  if (i >= n) return;
  int k = i % KK; int v = (i / KK) % VV; int b = i / (KK * VV);
  eT[(long long)b * KK * VV + (long long)k * VV + v] = f2bf(evecs[i]);
}

// weights (blk, rows, CW) -> transposed bf16 (blk, CW, rows)
__global__ void k_transpose_w(const float* __restrict__ in, unsigned short* __restrict__ outT,
                              int rows, long long n) {
  long long i = (long long)blockIdx.x * 256 + threadIdx.x;  // k fastest
  if (i >= n) return;
  int k   = (int)(i % rows);
  int c   = (int)((i / rows) % CW_);
  int blk = (int)(i / ((long long)rows * CW_));
  outT[i] = f2bf(in[(long long)blk * rows * CW_ + (long long)k * CW_ + c]);
}

// [A_re; -A_im] and [A_im; A_re] packed, stored transposed (CW x 2CW)
__global__ void k_pack(const float* __restrict__ Are, const float* __restrict__ Aim,
                       unsigned short* __restrict__ Wre2T, unsigned short* __restrict__ Wim2T, int n) {
  int i = blockIdx.x * 256 + threadIdx.x;  // over NB*CW*2CW, r fastest
  if (i >= n) return;
  int r  = i % (2 * CW_);
  int c  = (i / (2 * CW_)) % CW_;
  int ib = i / (2 * CW_ * CW_);
  const float* are = Are + (long long)ib * CW_ * CW_;
  const float* aim = Aim + (long long)ib * CW_ * CW_;
  float re, im;
  if (r < CW_) { re = are[r * CW_ + c];               im = aim[r * CW_ + c]; }
  else         { int q = r - CW_; re = -aim[q * CW_ + c]; im = are[q * CW_ + c]; }
  Wre2T[i] = f2bf(re); Wim2T[i] = f2bf(im);
}

__global__ void k_first(const float* __restrict__ x_in, const float* __restrict__ wf,
                        const float* __restrict__ bfirst, float* __restrict__ x, int n) {
  int i = blockIdx.x * 256 + threadIdx.x;
  if (i >= n) return;
  int c = i % CW_; long long bv = i / CW_;
  const float* xi = x_in + bv * CIN_;
  float s = bfirst[c];
  #pragma unroll
  for (int j = 0; j < CIN_; j++) s += xi[j] * wf[j * CW_ + c];
  x[i] = s;
}

// xm_t[b][c][v] = bf16(mass*x), comb[:,0:128] = bf16(x)   (v fastest)
__global__ void k_premul(const float* __restrict__ x, const float* __restrict__ mass,
                         unsigned short* __restrict__ xm_t, unsigned short* __restrict__ comb, int n) {
  int i = blockIdx.x * 256 + threadIdx.x;  // over B*CW*V
  if (i >= n) return;
  int v = i % VV; int c = (i / VV) % CW_; int b = i / (VV * CW_);
  long long bv = (long long)b * VV + v;
  float xv = x[bv * CW_ + c];
  xm_t[i] = f2bf(mass[bv] * xv);
  comb[bv * (3 * CW_) + c] = f2bf(xv);
}

// xsb_t[b][c][k] = bf16(exp(-eval*t) * x_spec[b][k][c])   (k fastest)
__global__ void k_spec(const float* __restrict__ xs, const float* __restrict__ evals,
                       const float* __restrict__ dt, unsigned short* __restrict__ xsb_t, int n) {
  int i = blockIdx.x * 256 + threadIdx.x;  // over B*CW*KK
  if (i >= n) return;
  int k = i % KK; int c = (i / KK) % CW_; int b = i / (KK * CW_);
  float t = fmaxf(dt[c], 1e-8f);
  float coef = __expf(-evals[b * KK + k] * t);
  xsb_t[i] = f2bf(coef * xs[((long long)b * KK + k) * CW_ + c]);
}

__global__ void k_tanh(const float* __restrict__ gx, const float* __restrict__ gy,
                       const float* __restrict__ vre, const float* __restrict__ vim,
                       unsigned short* __restrict__ comb, int n) {
  int i = blockIdx.x * 256 + threadIdx.x;
  if (i >= n) return;
  int c = i % CW_; long long bv = i / CW_;
  float g = tanhf(gx[i] * vre[i] + gy[i] * vim[i]);
  comb[bv * (3 * CW_) + 2 * CW_ + c] = f2bf(g);
}

__global__ void k_last(const float* __restrict__ x, const float* __restrict__ wl,
                       const float* __restrict__ bl, float* __restrict__ out, int n) {
  int i = blockIdx.x * 256 + threadIdx.x;
  if (i >= n) return;
  int o = i % COUT_; long long bv = i / COUT_;
  float s = bl[o];
  const float* xr = x + bv * CW_;
  #pragma unroll 8
  for (int c = 0; c < CW_; c++) s += xr[c] * wl[c * COUT_ + o];
  out[i] = s;
}

// ---------------------------------------------------------------------------

extern "C" void kernel_launch(void* const* d_in, const int* in_sizes, int n_in,
                              void* d_out, int out_size, void* d_ws, size_t ws_size,
                              hipStream_t stream) {
  const float* x_in    = (const float*)d_in[0];
  const float* mass    = (const float*)d_in[1];
  const float* evals   = (const float*)d_in[2];
  const float* evecs   = (const float*)d_in[3];
  const float* gradX   = (const float*)d_in[4];
  const float* gradY   = (const float*)d_in[5];
  const float* w_first = (const float*)d_in[6];
  const float* b_first = (const float*)d_in[7];
  const float* w_last  = (const float*)d_in[8];
  const float* b_last  = (const float*)d_in[9];
  const float* dt      = (const float*)d_in[10];
  const float* A_re    = (const float*)d_in[11];
  const float* A_im    = (const float*)d_in[12];
  const float* w0      = (const float*)d_in[13];
  const float* b0      = (const float*)d_in[14];
  const float* w1      = (const float*)d_in[15];
  const float* b1      = (const float*)d_in[16];
  const float* w2      = (const float*)d_in[17];
  const float* b2      = (const float*)d_in[18];
  float* out = (float*)d_out;

  char* ws = (char*)d_ws;
  size_t off = 0;
  auto alloc = [&](size_t bytes) -> char* {
    size_t r = off;
    off += (bytes + 255) & ~(size_t)255;
    return ws + r;
  };

  unsigned short* gXb    = (unsigned short*)alloc((size_t)BB * VV * VV * 2);
  unsigned short* gYb    = (unsigned short*)alloc((size_t)BB * VV * VV * 2);
  unsigned short* evb    = (unsigned short*)alloc((size_t)BB * VV * KK * 2);
  unsigned short* evTb   = (unsigned short*)alloc((size_t)BB * KK * VV * 2);
  float*          x      = (float*)         alloc((size_t)BB * VV * CW_ * 4);
  unsigned short* xm_t   = (unsigned short*)alloc((size_t)BB * CW_ * VV * 2);
  float*          xs     = (float*)         alloc((size_t)BB * KK * CW_ * 4);
  unsigned short* xsb_t  = (unsigned short*)alloc((size_t)BB * CW_ * KK * 2);
  unsigned short* comb   = (unsigned short*)alloc((size_t)BB * VV * 3 * CW_ * 2);
  unsigned short* xdifft = (unsigned short*)alloc((size_t)BB * CW_ * VV * 2);
  float*          gx     = (float*)         alloc((size_t)BB * VV * CW_ * 4);
  float*          gy     = (float*)         alloc((size_t)BB * VV * CW_ * 4);
  unsigned short* gxy    = (unsigned short*)alloc((size_t)BB * VV * 2 * CW_ * 2);
  float*          vre    = (float*)         alloc((size_t)BB * VV * CW_ * 4);
  float*          vim    = (float*)         alloc((size_t)BB * VV * CW_ * 4);
  unsigned short* h0     = (unsigned short*)alloc((size_t)BB * VV * CW_ * 2);
  unsigned short* h1     = (unsigned short*)alloc((size_t)BB * VV * CW_ * 2);
  unsigned short* w0t    = (unsigned short*)alloc((size_t)NB_ * 3 * CW_ * CW_ * 2);
  unsigned short* w1t    = (unsigned short*)alloc((size_t)NB_ * CW_ * CW_ * 2);
  unsigned short* w2t    = (unsigned short*)alloc((size_t)NB_ * CW_ * CW_ * 2);
  unsigned short* wre2t  = (unsigned short*)alloc((size_t)NB_ * 2 * CW_ * CW_ * 2);
  unsigned short* wim2t  = (unsigned short*)alloc((size_t)NB_ * 2 * CW_ * CW_ * 2);

  auto nblk = [](long long n) { return (unsigned)((n + 255) / 256); };
  const long long nGrad = (long long)BB * VV * VV;
  const long long nEv   = (long long)BB * VV * KK;
  const long long nBVC  = (long long)BB * VV * CW_;
  const long long nBKC  = (long long)BB * KK * CW_;

  // ---- one-time conversions / packing ----
  k_cvt<<<nblk(nGrad), 256, 0, stream>>>(gradX, gXb, nGrad);
  k_cvt<<<nblk(nGrad), 256, 0, stream>>>(gradY, gYb, nGrad);
  k_cvt<<<nblk(nEv),   256, 0, stream>>>(evecs, evb, nEv);
  k_transpose_cvt<<<nblk(nEv), 256, 0, stream>>>(evecs, evTb, (int)nEv);
  k_transpose_w<<<nblk((long long)NB_ * 3 * CW_ * CW_), 256, 0, stream>>>(w0, w0t, 3 * CW_, (long long)NB_ * 3 * CW_ * CW_);
  k_transpose_w<<<nblk((long long)NB_ * CW_ * CW_),     256, 0, stream>>>(w1, w1t, CW_, (long long)NB_ * CW_ * CW_);
  k_transpose_w<<<nblk((long long)NB_ * CW_ * CW_),     256, 0, stream>>>(w2, w2t, CW_, (long long)NB_ * CW_ * CW_);
  k_pack<<<nblk((long long)NB_ * 2 * CW_ * CW_), 256, 0, stream>>>(A_re, A_im, wre2t, wim2t, NB_ * 2 * CW_ * CW_);
  k_first<<<nblk(nBVC), 256, 0, stream>>>(x_in, w_first, b_first, x, (int)nBVC);

  const dim3 gV(VV / 128, 1, BB);
  const dim3 gKt(KK / 128, 1, BB);

  for (int i = 0; i < NB_; i++) {
    // mass*x -> xm_t (col-major bf16), comb[:,0:128] = x
    k_premul<<<nblk(nBVC), 256, 0, stream>>>(x, mass, xm_t, comb, (int)nBVC);

    // x_spec(K x C) = evecs^T @ (mass*x)  -> f32
    gemm_bf16<true,false,false,false,false,false><<<gKt, 256, 0, stream>>>(
        evTb, (long long)KK * VV, VV,
        xm_t, (long long)CW_ * VV, VV,
        xs,   (long long)KK * CW_, CW_,
        nullptr, 0, 0,  nullptr, 0, 0,
        nullptr, nullptr, 0, 0,
        KK, VV);

    // coefs * x_spec -> xsb_t (col-major bf16)
    k_spec<<<nblk(nBKC), 256, 0, stream>>>(xs, evals, dt + i * CW_, xsb_t, (int)nBKC);

    // x_diff = evecs @ (coefs*x_spec) -> comb[:,128:256] (row) + xdifft (col)
    gemm_bf16<false,true,true,false,false,false><<<gV, 256, 0, stream>>>(
        evb,   (long long)VV * KK, KK,
        xsb_t, (long long)CW_ * KK, KK,
        nullptr, 0, 0,
        comb + CW_, (long long)VV * 3 * CW_, 3 * CW_,
        xdifft, (long long)CW_ * VV, VV,
        nullptr, nullptr, 0, 0,
        VV, KK);

    // gx = gradX @ x_diff  (f32 + packed bf16 [gx|gy])
    gemm_bf16<true,true,false,false,false,false><<<gV, 256, 0, stream>>>(
        gXb, (long long)VV * VV, VV,
        xdifft, (long long)CW_ * VV, VV,
        gx, (long long)VV * CW_, CW_,
        gxy, (long long)VV * 2 * CW_, 2 * CW_,
        nullptr, 0, 0,
        nullptr, nullptr, 0, 0,
        VV, VV);
    // gy = gradY @ x_diff
    gemm_bf16<true,true,false,false,false,false><<<gV, 256, 0, stream>>>(
        gYb, (long long)VV * VV, VV,
        xdifft, (long long)CW_ * VV, VV,
        gy, (long long)VV * CW_, CW_,
        gxy + CW_, (long long)VV * 2 * CW_, 2 * CW_,
        nullptr, 0, 0,
        nullptr, nullptr, 0, 0,
        VV, VV);

    // vBre = [gx gy] @ [A_re; -A_im],  vBim = [gx gy] @ [A_im; A_re]
    gemm_bf16<true,false,false,false,false,false><<<gV, 256, 0, stream>>>(
        gxy, (long long)VV * 2 * CW_, 2 * CW_,
        wre2t + (long long)i * 2 * CW_ * CW_, 0, 2 * CW_,
        vre, (long long)VV * CW_, CW_,
        nullptr, 0, 0,  nullptr, 0, 0,
        nullptr, nullptr, 0, 0,
        VV, 2 * CW_);
    gemm_bf16<true,false,false,false,false,false><<<gV, 256, 0, stream>>>(
        gxy, (long long)VV * 2 * CW_, 2 * CW_,
        wim2t + (long long)i * 2 * CW_ * CW_, 0, 2 * CW_,
        vim, (long long)VV * CW_, CW_,
        nullptr, 0, 0,  nullptr, 0, 0,
        nullptr, nullptr, 0, 0,
        VV, 2 * CW_);

    // grad_feat = tanh(gx*vBre + gy*vBim) -> comb[:,256:384]
    k_tanh<<<nblk(nBVC), 256, 0, stream>>>(gx, gy, vre, vim, comb, (int)nBVC);

    // h0 = relu(comb @ w0_i + b0_i) -> bf16
    gemm_bf16<false,true,false,true,false,true><<<gV, 256, 0, stream>>>(
        comb, (long long)VV * 3 * CW_, 3 * CW_,
        w0t + (long long)i * 3 * CW_ * CW_, 0, 3 * CW_,
        nullptr, 0, 0,
        h0, (long long)VV * CW_, CW_,
        nullptr, 0, 0,
        b0 + i * CW_, nullptr, 0, 0,
        VV, 3 * CW_);
    // h1 = relu(h0 @ w1_i + b1_i) -> bf16
    gemm_bf16<false,true,false,true,false,true><<<gV, 256, 0, stream>>>(
        h0, (long long)VV * CW_, CW_,
        w1t + (long long)i * CW_ * CW_, 0, CW_,
        nullptr, 0, 0,
        h1, (long long)VV * CW_, CW_,
        nullptr, 0, 0,
        b1 + i * CW_, nullptr, 0, 0,
        VV, CW_);
    // x = h1 @ w2_i + b2_i + x   (residual, f32)
    gemm_bf16<true,false,false,true,true,false><<<gV, 256, 0, stream>>>(
        h1, (long long)VV * CW_, CW_,
        w2t + (long long)i * CW_ * CW_, 0, CW_,
        x, (long long)VV * CW_, CW_,
        nullptr, 0, 0,  nullptr, 0, 0,
        b2 + i * CW_, x, (long long)VV * CW_, CW_,
        VV, CW_);
  }

  // out = x @ w_last + b_last
  k_last<<<nblk((long long)BB * VV * COUT_), 256, 0, stream>>>(x, w_last, b_last, out, BB * VV * COUT_);
}
